// GIN_52115133169838
// MI455X (gfx1250) — matrix operand order repaired and checked by reading, not compiled, not run on hardware
//
#include <hip/hip_runtime.h>
#include <hip/hip_bf16.h>

// ---------------------------------------------------------------------------
// CDNA5 WMMA types (wave32): v16bf A/B fragments, v8f C/D accumulator
// ---------------------------------------------------------------------------
typedef __attribute__((ext_vector_type(16))) __bf16 v16bf;
typedef __attribute__((ext_vector_type(8)))  float  v8f;
union ABu { v16bf v; unsigned short u[16]; };

__device__ __forceinline__ unsigned short f2bf(float f) {
  unsigned int x = __float_as_uint(f);
  x += 0x7FFFu + ((x >> 16) & 1u);          // round-to-nearest-even
  return (unsigned short)(x >> 16);
}

// A fragment: 16x32 bf16 tile, lane = (n, hi). Contiguous K order (matches ldB):
//  e<8  -> K = k0 + hi*16 + e
//  e>=8 -> K = k0 + hi*16 + 8 + (e-8)
__device__ __forceinline__ v16bf ldA(const unsigned short* __restrict__ row, int k0, int hi) {
  ABu a;
#pragma unroll
  for (int e = 0; e < 8; ++e) a.u[e]     = row[k0 + hi * 16 + e];
#pragma unroll
  for (int e = 0; e < 8; ++e) a.u[8 + e] = row[k0 + hi * 16 + 8 + e];
  return a.v;
}

// B fragment: 32x16 bf16 tile. Lane (n,hi) holds column n, K = k0 + hi*16 + e.
// With row-major weights W[j][k] this is 16 CONTIGUOUS bf16 of row j.
__device__ __forceinline__ v16bf ldB(const unsigned short* __restrict__ p) {
  ABu b;
#pragma unroll
  for (int e = 0; e < 16; ++e) b.u[e] = p[e];
  return b.v;
}

// ---------------------------------------------------------------------------
// Small helper kernels
// ---------------------------------------------------------------------------
__global__ void k_tobf16(const float* __restrict__ w, unsigned short* __restrict__ o, int n) {
  int i = blockIdx.x * 256 + threadIdx.x;
  if (i < n) o[i] = f2bf(w[i]);
}

// h0[n][:] = emb[x[n]][:]  (one float4 per thread)
__global__ void k_gather(const int* __restrict__ x, const float* __restrict__ emb,
                         float* __restrict__ h0, int n) {
  int t = blockIdx.x * 256 + threadIdx.x;
  int node = t >> 5, q = (t & 31) << 2;
  if (node >= n) return;
  float4 v = *(const float4*)(emb + x[node] * 128 + q);
  *(float4*)(h0 + node * 128 + q) = v;
}

// m = relu(h[src] + edge_attr*ew + eb); aggr[dst] += m   (L2-resident scatter)
__global__ void k_edge(const int* __restrict__ ei, const float* __restrict__ eattr,
                       const float* __restrict__ ew, const float* __restrict__ eb,
                       const float* __restrict__ hsrc, float* __restrict__ aggr, int E) {
  int t = blockIdx.x * 256 + threadIdx.x;
  int e = t >> 5, q = (t & 31) << 2;
  if (e >= E) return;
  int s = ei[e];
  int d = ei[E + e];
  float w = eattr[e];
  float4 h  = *(const float4*)(hsrc + s * 128 + q);
  float4 cw = *(const float4*)(ew + q);
  float4 cb = *(const float4*)(eb + q);
  float m0 = fmaxf(h.x + fmaf(w, cw.x, cb.x), 0.f);
  float m1 = fmaxf(h.y + fmaf(w, cw.y, cb.y), 0.f);
  float m2 = fmaxf(h.z + fmaf(w, cw.z, cb.z), 0.f);
  float m3 = fmaxf(h.w + fmaf(w, cw.w, cb.w), 0.f);
  float* a = aggr + d * 128 + q;
  atomicAdd(a + 0, m0);
  atomicAdd(a + 1, m1);
  atomicAdd(a + 2, m2);
  atomicAdd(a + 3, m3);
}

// mu/var -> fused affine:  y = t*affA + affB  ==  g*(t-mu)*rsqrt(var+eps)+be
__global__ void k_bnfin(const float* __restrict__ ssum, const float* __restrict__ ssq,
                        const float* __restrict__ g, const float* __restrict__ be,
                        float* __restrict__ affA, float* __restrict__ affB, float invN) {
  int j = threadIdx.x;
  float mu  = ssum[j] * invN;
  float var = ssq[j] * invN - mu * mu;
  float rs  = rsqrtf(var + 1e-5f);
  float a   = g[j] * rs;
  affA[j] = a;
  affB[j] = fmaf(-mu, a, be[j]);
}

// ---------------------------------------------------------------------------
// WMMA GEMM: OUT[n][j] = IN[n][:] @ W[j][:] + bias[j], K = 128, 16-row tiles.
// Block = 256 threads = 8 waves; wave w owns columns [16w, 16w+16).
// MODE 0: write raw pre-activation + accumulate per-column sum / sumsq (BN)
// MODE 1: A-tile gets affine+relu applied (BN), output relu'd + graph-pooled
// ---------------------------------------------------------------------------
template <int MODE>
__global__ void k_mm128(const float* __restrict__ IN, const unsigned short* __restrict__ W,
                        const float* __restrict__ bias, float* __restrict__ OUT,
                        float* __restrict__ ssum, float* __restrict__ ssq,
                        const float* __restrict__ affA, const float* __restrict__ affB,
                        const int* __restrict__ batch, float* __restrict__ pool, int nrows) {
  __shared__ unsigned short sA[16 * 128];
  const int row0 = blockIdx.x * 16;
  const int tid  = threadIdx.x;
  for (int i = tid; i < 16 * 128; i += 256) {
    int r = i >> 7, cidx = i & 127;
    int gr = row0 + r;
    float f = (gr < nrows) ? IN[gr * 128 + cidx] : 0.f;
    if (MODE == 1) f = fmaxf(fmaf(f, affA[cidx], affB[cidx]), 0.f);
    sA[i] = f2bf(f);
  }
  __syncthreads();

  const int wave = tid >> 5, lane = tid & 31;
  const int n = lane & 15, hi = lane >> 4;
  const int j = wave * 16 + n;
  float bv = bias[j];
  v8f c = {bv, bv, bv, bv, bv, bv, bv, bv};
  const unsigned short* arow = sA + n * 128;
#pragma unroll
  for (int k0 = 0; k0 < 128; k0 += 32) {
    v16bf a = ldA(arow, k0, hi);
    v16bf b = ldB(W + j * 128 + k0 + hi * 16);
    c = __builtin_amdgcn_wmma_f32_16x16x32_bf16(false, a, false, b, (short)0, c, false, false);
  }

  if (MODE == 0) {
    float s = 0.f, s2 = 0.f;
#pragma unroll
    for (int r = 0; r < 8; ++r) {
      int gr = row0 + hi * 8 + r;          // D layout: M = r + 8*hi, N = n
      if (gr < nrows) {
        float v = c[r];
        OUT[gr * 128 + j] = v;
        s += v; s2 += v * v;
      }
    }
    atomicAdd(ssum + j, s);
    atomicAdd(ssq  + j, s2);
  } else {
#pragma unroll
    for (int r = 0; r < 8; ++r) {
      int gr = row0 + hi * 8 + r;
      if (gr < nrows) {
        float v = fmaxf(c[r], 0.f);
        OUT[gr * 128 + j] = v;
        atomicAdd(pool + batch[gr] * 128 + j, v);  // fused segment_sum over graphs
      }
    }
  }
}

// ---------------------------------------------------------------------------
// Readout head: [p1|p2] (Gx256) -> relu(@W1^T+B1) (Gx128) -> @W2^T+B2 (GxT)
// Block = 16 graphs, 8 waves. Intermediate kept in LDS as bf16.
// ---------------------------------------------------------------------------
__global__ void k_head(const float* __restrict__ p1, const float* __restrict__ p2,
                       const unsigned short* __restrict__ W1, const float* __restrict__ B1,
                       const unsigned short* __restrict__ W2, const float* __restrict__ B2,
                       float* __restrict__ out, int T) {
  __shared__ unsigned short sA[16 * 256];
  __shared__ unsigned short sH[16 * 128];
  const int row0 = blockIdx.x * 16;
  const int tid  = threadIdx.x;
  for (int i = tid; i < 16 * 256; i += 256) {
    int r = i >> 8, cidx = i & 255;
    int g = row0 + r;
    float f = (cidx < 128) ? p1[g * 128 + cidx] : p2[g * 128 + (cidx - 128)];
    sA[i] = f2bf(f);
  }
  __syncthreads();

  const int wave = tid >> 5, lane = tid & 31;
  const int n = lane & 15, hi = lane >> 4;

  {  // layer 1: K=256 -> 128 cols, relu
    const int j = wave * 16 + n;
    float bv = B1[j];
    v8f c = {bv, bv, bv, bv, bv, bv, bv, bv};
    const unsigned short* arow = sA + n * 256;
#pragma unroll
    for (int k0 = 0; k0 < 256; k0 += 32) {
      v16bf a = ldA(arow, k0, hi);
      v16bf b = ldB(W1 + j * 256 + k0 + hi * 16);
      c = __builtin_amdgcn_wmma_f32_16x16x32_bf16(false, a, false, b, (short)0, c, false, false);
    }
#pragma unroll
    for (int r = 0; r < 8; ++r) sH[(hi * 8 + r) * 128 + j] = f2bf(fmaxf(c[r], 0.f));
  }
  __syncthreads();

  // layer 2: K=128 -> T cols (T/128 tiles per wave)
  const int tiles = T / 128;
  for (int jt = 0; jt < tiles; ++jt) {
    const int j = (wave * tiles + jt) * 16 + n;
    float bv = B2[j];
    v8f c = {bv, bv, bv, bv, bv, bv, bv, bv};
    const unsigned short* arow = sH + n * 128;
#pragma unroll
    for (int k0 = 0; k0 < 128; k0 += 32) {
      v16bf a = ldA(arow, k0, hi);
      v16bf b = ldB(W2 + j * 128 + k0 + hi * 16);
      c = __builtin_amdgcn_wmma_f32_16x16x32_bf16(false, a, false, b, (short)0, c, false, false);
    }
#pragma unroll
    for (int r = 0; r < 8; ++r) out[(row0 + hi * 8 + r) * T + j] = c[r];
  }
}

// ---------------------------------------------------------------------------
extern "C" void kernel_launch(void* const* d_in, const int* in_sizes, int n_in,
                              void* d_out, int out_size, void* d_ws, size_t ws_size,
                              hipStream_t stream) {
  const int*   x     = (const int*)  d_in[0];
  const int*   ei    = (const int*)  d_in[1];
  const float* eattr = (const float*)d_in[2];
  const int*   batch = (const int*)  d_in[3];
  const float* emb   = (const float*)d_in[4];
  const float* e1_w  = (const float*)d_in[5];
  const float* e1_b  = (const float*)d_in[6];
  const float* c1_w1 = (const float*)d_in[7];
  const float* c1_b1 = (const float*)d_in[8];
  const float* c1_g  = (const float*)d_in[9];
  const float* c1_be = (const float*)d_in[10];
  const float* c1_w2 = (const float*)d_in[11];
  const float* c1_b2 = (const float*)d_in[12];
  const float* e2_w  = (const float*)d_in[13];
  const float* e2_b  = (const float*)d_in[14];
  const float* c2_w1 = (const float*)d_in[15];
  const float* c2_b1 = (const float*)d_in[16];
  const float* c2_g  = (const float*)d_in[17];
  const float* c2_be = (const float*)d_in[18];
  const float* c2_w2 = (const float*)d_in[19];
  const float* c2_b2 = (const float*)d_in[20];
  const float* f_w1  = (const float*)d_in[21];
  const float* f_b1  = (const float*)d_in[22];
  const float* f_w2  = (const float*)d_in[23];
  const float* f_b2  = (const float*)d_in[24];
  const float* b_w1  = (const float*)d_in[25];
  const float* b_b1  = (const float*)d_in[26];
  const float* b_w2  = (const float*)d_in[27];
  const float* b_b2  = (const float*)d_in[28];

  const int N = in_sizes[0];          // 100000
  const int E = in_sizes[2];          // 1600000
  const int T = in_sizes[24];         // 512
  const int G = out_size / (2 * T);   // 2048

  // ---- workspace carve-out (all buffers written before read each launch) ---
  char* wsp = (char*)d_ws;
  size_t off = 0;
  auto take = [&](size_t bytes) -> void* {
    void* p = wsp + off;
    off += (bytes + 255) & ~(size_t)255;
    return p;
  };
  float* buf0 = (float*)take((size_t)N * 128 * 4);
  float* buf1 = (float*)take((size_t)N * 128 * 4);
  float* buf2 = (float*)take((size_t)N * 128 * 4);
  float* p1   = (float*)take((size_t)G * 128 * 4);
  float* p2   = (float*)take((size_t)G * 128 * 4);
  float* ssum = (float*)take(128 * 4);
  float* ssq  = (float*)take(128 * 4);
  float* affA = (float*)take(128 * 4);
  float* affB = (float*)take(128 * 4);
  unsigned short* wb_c1w1 = (unsigned short*)take(128 * 128 * 2);
  unsigned short* wb_c1w2 = (unsigned short*)take(128 * 128 * 2);
  unsigned short* wb_c2w1 = (unsigned short*)take(128 * 128 * 2);
  unsigned short* wb_c2w2 = (unsigned short*)take(128 * 128 * 2);
  unsigned short* wb_fw1  = (unsigned short*)take(128 * 256 * 2);
  unsigned short* wb_fw2  = (unsigned short*)take((size_t)T * 128 * 2);
  unsigned short* wb_bw1  = (unsigned short*)take(128 * 256 * 2);
  unsigned short* wb_bw2  = (unsigned short*)take((size_t)T * 128 * 2);

  auto cvt = [&](const float* src, unsigned short* dst, int n) {
    k_tobf16<<<(n + 255) / 256, 256, 0, stream>>>(src, dst, n);
  };
  cvt(c1_w1, wb_c1w1, 128 * 128);
  cvt(c1_w2, wb_c1w2, 128 * 128);
  cvt(c2_w1, wb_c2w1, 128 * 128);
  cvt(c2_w2, wb_c2w2, 128 * 128);
  cvt(f_w1,  wb_fw1,  128 * 256);
  cvt(f_w2,  wb_fw2,  T * 128);
  cvt(b_w1,  wb_bw1,  128 * 256);
  cvt(b_w2,  wb_bw2,  T * 128);

  hipMemsetAsync(p1, 0, (size_t)G * 128 * 4, stream);
  hipMemsetAsync(p2, 0, (size_t)G * 128 * 4, stream);

  const int gatherBlocks = (N * 32 + 255) / 256;
  const int edgeBlocks   = (E * 32 + 255) / 256;
  const int mmBlocks     = (N + 15) / 16;

  // h0 = emb[x]
  k_gather<<<gatherBlocks, 256, 0, stream>>>(x, emb, buf0, N);

  // ---- GINE layer 1 ----
  hipMemcpyAsync(buf1, buf0, (size_t)N * 128 * 4, hipMemcpyDeviceToDevice, stream); // aggr = h0
  k_edge<<<edgeBlocks, 256, 0, stream>>>(ei, eattr, e1_w, e1_b, buf0, buf1, E);     // += relu msgs
  hipMemsetAsync(ssum, 0, 128 * 4, stream);
  hipMemsetAsync(ssq,  0, 128 * 4, stream);
  k_mm128<0><<<mmBlocks, 256, 0, stream>>>(buf1, wb_c1w1, c1_b1, buf2,
                                           ssum, ssq, nullptr, nullptr, nullptr, nullptr, N);
  k_bnfin<<<1, 128, 0, stream>>>(ssum, ssq, c1_g, c1_be, affA, affB, 1.0f / (float)N);
  k_mm128<1><<<mmBlocks, 256, 0, stream>>>(buf2, wb_c1w2, c1_b2, buf1,
                                           nullptr, nullptr, affA, affB, batch, p1, N); // h1 + pool

  // ---- GINE layer 2 ----
  hipMemcpyAsync(buf0, buf1, (size_t)N * 128 * 4, hipMemcpyDeviceToDevice, stream); // aggr = h1
  k_edge<<<edgeBlocks, 256, 0, stream>>>(ei, eattr, e2_w, e2_b, buf1, buf0, E);
  hipMemsetAsync(ssum, 0, 128 * 4, stream);
  hipMemsetAsync(ssq,  0, 128 * 4, stream);
  k_mm128<0><<<mmBlocks, 256, 0, stream>>>(buf0, wb_c2w1, c2_b1, buf2,
                                           ssum, ssq, nullptr, nullptr, nullptr, nullptr, N);
  k_bnfin<<<1, 128, 0, stream>>>(ssum, ssq, c2_g, c2_be, affA, affB, 1.0f / (float)N);
  k_mm128<1><<<mmBlocks, 256, 0, stream>>>(buf2, wb_c2w2, c2_b2, buf0,
                                           nullptr, nullptr, affA, affB, batch, p2, N); // h2 + pool

  // ---- heads ----
  float* outF = (float*)d_out;
  float* outB = outF + (size_t)G * T;
  k_head<<<G / 16, 256, 0, stream>>>(p1, p2, wb_fw1, f_b1, wb_fw2, f_b2, outF, T);
  k_head<<<G / 16, 256, 0, stream>>>(p1, p2, wb_bw1, b_b1, wb_bw2, b_b2, outB, T);
}
